// MaxFlow_47287589929094
// MI455X (gfx1250) — matrix operand order repaired and checked by reading, not compiled, not run on hardware
//
#include <hip/hip_runtime.h>
#include <math.h>

// Problem constants (match reference)
#define ME 16384      // M_EDGES: columns of A, length of x/b/c
#define NR 4094       // valid rows of A (N_NODES-2)
#define NP 4096       // padded rows (multiple of 16)
#define KSPLIT 8      // K-dimension split for matvec parallelism
#define PDHG_ITERS 300
#define POWER_ITERS 20

typedef __attribute__((ext_vector_type(16))) __bf16 bf16x16;
typedef __attribute__((ext_vector_type(8)))  __bf16 bf16x8;
typedef __attribute__((ext_vector_type(8)))  float  floatx8;

// ---------------------------------------------------------------------------
// Setup kernels
// ---------------------------------------------------------------------------

// A (NR x ME f32) -> Abf (NP x ME bf16), zero-padding rows NR..NP-1.
__global__ void mf_convert_pad(const float* __restrict__ A,
                               __bf16* __restrict__ Abf) {
  size_t idx = (size_t)blockIdx.x * blockDim.x + threadIdx.x;  // < NP*ME
  int row = (int)(idx >> 14);                                  // ME == 2^14
  float v = 0.0f;
  if (row < NR) v = A[idx];
  Abf[idx] = (__bf16)v;
}

// Abf (NP x ME) -> ATbf (ME x NP), 32x32 LDS tiles.
__global__ void mf_transpose(const __bf16* __restrict__ in,
                             __bf16* __restrict__ out) {
  __shared__ __bf16 tile[32][33];
  int c0 = blockIdx.x * 32;   // column block in Abf (0..ME)
  int r0 = blockIdx.y * 32;   // row block in Abf    (0..NP)
  for (int i = threadIdx.y; i < 32; i += 8)
    tile[i][threadIdx.x] = in[(size_t)(r0 + i) * ME + c0 + threadIdx.x];
  __syncthreads();
  for (int i = threadIdx.y; i < 32; i += 8)
    out[(size_t)(c0 + i) * NP + r0 + threadIdx.x] = tile[threadIdx.x][i];
}

// x=0, y=0, ybf=0, vbf = 1/sqrt(ME)
__global__ void mf_init(float* __restrict__ x, float* __restrict__ y,
                        __bf16* __restrict__ ybf, __bf16* __restrict__ vbf) {
  int i = blockIdx.x * blockDim.x + threadIdx.x;
  if (i < ME) { x[i] = 0.0f; vbf[i] = (__bf16)0.0078125f; }  // 1/128
  if (i < NP) { y[i] = 0.0f; ybf[i] = (__bf16)0.0f; }
}

// ---------------------------------------------------------------------------
// WMMA bf16 matvec:  part[split][row] = sum_{k in split} Abf[row][k] * v[k]
// One wave computes 16 output rows; vector is replicated across the 16 B
// columns so every D column holds the matvec result (redundancy is free:
// memory-bound). A-operand: lanes 0-15 hold K=0..15, lanes 16-31 hold K=16..31 (as B);
// B-operand: lanes 0-15 hold K=0..15, lanes 16-31 hold K=16..31.
// K-loop unrolled 4x so 4 load groups are in flight per wave before the
// first s_wait_loadcnt (latency hiding for the 23.3 TB/s stream).
// ---------------------------------------------------------------------------
__global__ __launch_bounds__(256)
void mf_wmma_matvec(const __bf16* __restrict__ Abf,
                    const __bf16* __restrict__ vbf,
                    float* __restrict__ part,
                    int M, int K, int kPerSplit) {
  const int lane = threadIdx.x & 31;
  const int waveInBlock = threadIdx.x >> 5;
  const int row0 = (blockIdx.x * (blockDim.x >> 5) + waveInBlock) * 16;
  if (row0 >= M) return;
  const int kBeg = blockIdx.y * kPerSplit;
  const int kEnd = kBeg + kPerSplit;
  const int m  = lane & 15;
  const int hi = lane >> 4;  // 0: K 0..15, 1: K 16..31
  const __bf16* arow = Abf + (size_t)(row0 + m) * K + (size_t)hi * 16;

  floatx8 acc = {0.f, 0.f, 0.f, 0.f, 0.f, 0.f, 0.f, 0.f};
  // kPerSplit is a multiple of 128 (512 or 2048).
  for (int k0 = kBeg; k0 < kEnd; k0 += 128) {
    __builtin_prefetch((const void*)(arow + k0 + 1024), 0, 3);  // 2KB ahead
#pragma unroll
    for (int u = 0; u < 4; ++u) {
      const int k = k0 + u * 32;
      bf16x8 a_lo = *(const bf16x8*)(arow + k);        // halves 0..7
      bf16x8 a_hi = *(const bf16x8*)(arow + k + 8);    // halves 8..15
      bf16x16 aop = __builtin_shufflevector(a_lo, a_hi,
          0, 1, 2, 3, 4, 5, 6, 7, 8, 9, 10, 11, 12, 13, 14, 15);
      bf16x16 bop = *(const bf16x16*)(vbf + k + hi * 16);
      acc = __builtin_amdgcn_wmma_f32_16x16x32_bf16(
          /*neg_a=*/false, aop, /*neg_b=*/false, bop,
          /*c_mod=*/(short)0, acc, /*reuse_a=*/false, /*reuse_b=*/false);
    }
  }
  // D layout: VGPR r, lanes 0-15 -> M=r (N=lane), lanes 16-31 -> M=r+8.
  // All N columns are identical; lane 0 and lane 16 hold the 16 rows.
  if (m == 0) {
    float* o = part + (size_t)blockIdx.y * M + row0 + hi * 8;
#pragma unroll
    for (int r = 0; r < 8; ++r) o[r] = acc[r];
  }
}

// ---------------------------------------------------------------------------
// Partial-sum consumers / pointwise kernels
// ---------------------------------------------------------------------------

__global__ void mf_sum_part_bf16(const float* __restrict__ part, int M,
                                 __bf16* __restrict__ out) {
  int i = blockIdx.x * blockDim.x + threadIdx.x;
  if (i < M) {
    float s = 0.0f;
#pragma unroll
    for (int k = 0; k < KSPLIT; ++k) s += part[k * M + i];
    out[i] = (__bf16)s;
  }
}

__global__ void mf_sumsq_part(const float* __restrict__ part, int M,
                              float* __restrict__ out) {
  __shared__ float sm[256];
  float s = 0.0f;
  for (int i = threadIdx.x; i < M; i += 256) {
    float v = 0.0f;
#pragma unroll
    for (int k = 0; k < KSPLIT; ++k) v += part[k * M + i];
    s += v * v;
  }
  sm[threadIdx.x] = s;
  __syncthreads();
  for (int o = 128; o > 0; o >>= 1) {
    if ((int)threadIdx.x < o) sm[threadIdx.x] += sm[threadIdx.x + o];
    __syncthreads();
  }
  if (threadIdx.x == 0) out[0] = sm[0];
}

__global__ void mf_normalize_bf16(const float* __restrict__ part, int M,
                                  const float* __restrict__ sumsq,
                                  __bf16* __restrict__ out) {
  int i = blockIdx.x * blockDim.x + threadIdx.x;
  if (i < M) {
    float s = 0.0f;
#pragma unroll
    for (int k = 0; k < KSPLIT; ++k) s += part[k * M + i];
    out[i] = (__bf16)(s * rsqrtf(sumsq[0]));
  }
}

// nrm = sqrt(||A^T A v||) = sumsq^(1/4);  tau = sigma = 0.9 / nrm
__global__ void mf_steps(const float* __restrict__ sumsq,
                         float* __restrict__ step) {
  if (threadIdx.x == 0) step[0] = 0.9f / sqrtf(sqrtf(sumsq[0]));
}

// x_new = clip(x - tau*(c + A^T y), 0, b); zbf = bf16(2*x_new - x_old)
__global__ void mf_update_x(float* __restrict__ x,
                            const float* __restrict__ c,
                            const float* __restrict__ b,
                            const float* __restrict__ atyPart,
                            const float* __restrict__ step,
                            __bf16* __restrict__ zbf) {
  int i = blockIdx.x * blockDim.x + threadIdx.x;
  if (i < ME) {
    float aty = 0.0f;
#pragma unroll
    for (int k = 0; k < KSPLIT; ++k) aty += atyPart[k * ME + i];
    float tau = step[0];
    float xo = x[i];
    float xn = xo - tau * (c[i] + aty);
    xn = fminf(fmaxf(xn, 0.0f), b[i]);
    x[i] = xn;
    zbf[i] = (__bf16)(2.0f * xn - xo);
  }
}

// y += sigma * (A z); ybf = bf16(y)   (padded rows stay 0: A pad rows are 0)
__global__ void mf_update_y(float* __restrict__ y,
                            const float* __restrict__ azPart,
                            const float* __restrict__ step,
                            __bf16* __restrict__ ybf) {
  int i = blockIdx.x * blockDim.x + threadIdx.x;
  if (i < NP) {
    float az = 0.0f;
#pragma unroll
    for (int k = 0; k < KSPLIT; ++k) az += azPart[k * NP + i];
    float yn = y[i] + step[0] * az;
    y[i] = yn;
    ybf[i] = (__bf16)yn;
  }
}

// out[0] = sum(x * -c)
__global__ void mf_objective(const float* __restrict__ x,
                             const float* __restrict__ c,
                             float* __restrict__ out) {
  __shared__ float sm[256];
  float s = 0.0f;
  for (int i = threadIdx.x; i < ME; i += 256) s += x[i] * (-c[i]);
  sm[threadIdx.x] = s;
  __syncthreads();
  for (int o = 128; o > 0; o >>= 1) {
    if ((int)threadIdx.x < o) sm[threadIdx.x] += sm[threadIdx.x + o];
    __syncthreads();
  }
  if (threadIdx.x == 0) out[0] = sm[0];
}

// ---------------------------------------------------------------------------
// Host driver
// ---------------------------------------------------------------------------
extern "C" void kernel_launch(void* const* d_in, const int* in_sizes, int n_in,
                              void* d_out, int out_size, void* d_ws, size_t ws_size,
                              hipStream_t stream) {
  const float* A = (const float*)d_in[0];  // (NR, ME)
  const float* b = (const float*)d_in[1];  // (ME,)
  const float* c = (const float*)d_in[2];  // (ME,)

  // Carve workspace (256B aligned slices).
  size_t off = 0;
  auto carve = [&](size_t bytes) {
    void* p = (char*)d_ws + off;
    off += (bytes + 255) & ~(size_t)255;
    return p;
  };
  __bf16* Abf   = (__bf16*)carve((size_t)NP * ME * 2);   // 128 MiB, row-major
  __bf16* ATbf  = (__bf16*)carve((size_t)ME * NP * 2);   // 128 MiB, row-major
  float*  x     = (float*) carve((size_t)ME * 4);
  __bf16* zbf   = (__bf16*)carve((size_t)ME * 2);
  float*  y     = (float*) carve((size_t)NP * 4);
  __bf16* ybf   = (__bf16*)carve((size_t)NP * 2);
  __bf16* vbf   = (__bf16*)carve((size_t)ME * 2);
  __bf16* avbf  = (__bf16*)carve((size_t)NP * 2);
  float*  partB = (float*) carve((size_t)KSPLIT * ME * 4);  // M=ME outputs
  float*  partS = (float*) carve((size_t)KSPLIT * NP * 4);  // M=NP outputs
  float*  sumsq = (float*) carve(256);
  float*  step  = (float*) carve(256);
  (void)ws_size; (void)in_sizes; (void)n_in; (void)out_size;

  const dim3 blk256(256);
  const dim3 gridMvA(NP / 128, KSPLIT);   // A:  M=NP,  K=ME
  const dim3 gridMvAT(ME / 128, KSPLIT);  // AT: M=ME,  K=NP
  const int  kpsA  = ME / KSPLIT;         // 2048
  const int  kpsAT = NP / KSPLIT;         // 512

  // Setup: bf16 conversion (pad), transpose, state init.
  mf_convert_pad<<<(NP * (size_t)ME) / 256, blk256, 0, stream>>>(A, Abf);
  mf_transpose<<<dim3(ME / 32, NP / 32), dim3(32, 8), 0, stream>>>(Abf, ATbf);
  mf_init<<<ME / 256, blk256, 0, stream>>>(x, y, ybf, vbf);

  // Power iteration: v <- normalize(A^T (A v))
  for (int it = 0; it < POWER_ITERS; ++it) {
    mf_wmma_matvec<<<gridMvA, blk256, 0, stream>>>(Abf, vbf, partS, NP, ME, kpsA);
    mf_sum_part_bf16<<<NP / 256, blk256, 0, stream>>>(partS, NP, avbf);
    mf_wmma_matvec<<<gridMvAT, blk256, 0, stream>>>(ATbf, avbf, partB, ME, NP, kpsAT);
    mf_sumsq_part<<<1, blk256, 0, stream>>>(partB, ME, sumsq);
    mf_normalize_bf16<<<ME / 256, blk256, 0, stream>>>(partB, ME, sumsq, vbf);
  }
  // nrm = sqrt(||A^T (A v)||)  ->  tau = sigma = 0.9/nrm
  mf_wmma_matvec<<<gridMvA, blk256, 0, stream>>>(Abf, vbf, partS, NP, ME, kpsA);
  mf_sum_part_bf16<<<NP / 256, blk256, 0, stream>>>(partS, NP, avbf);
  mf_wmma_matvec<<<gridMvAT, blk256, 0, stream>>>(ATbf, avbf, partB, ME, NP, kpsAT);
  mf_sumsq_part<<<1, blk256, 0, stream>>>(partB, ME, sumsq);
  mf_steps<<<1, 32, 0, stream>>>(sumsq, step);

  // PDHG iterations.
  for (int it = 0; it < PDHG_ITERS; ++it) {
    mf_wmma_matvec<<<gridMvAT, blk256, 0, stream>>>(ATbf, ybf, partB, ME, NP, kpsAT);
    mf_update_x<<<ME / 256, blk256, 0, stream>>>(x, c, b, partB, step, zbf);
    mf_wmma_matvec<<<gridMvA, blk256, 0, stream>>>(Abf, zbf, partS, NP, ME, kpsA);
    mf_update_y<<<NP / 256, blk256, 0, stream>>>(y, partS, step, ybf);
  }

  // Objective: sum(x * -c)
  mf_objective<<<1, blk256, 0, stream>>>(x, c, (float*)d_out);
}
